// SMG_51161650430431
// MI455X (gfx1250) — matrix operand order repaired and checked by reading, not compiled, not run on hardware
//
#include <hip/hip_runtime.h>

#define HIDDEN   128
#define AOUT     64
#define NGRAPHS  256
#define INFEAT   70
#define NEG_SLOPE 0.2f

typedef __attribute__((ext_vector_type(16))) _Float16 v16h;
typedef __attribute__((ext_vector_type(8)))  float    v8f;

static inline unsigned gdim(long long total, int block) {
  return (unsigned)((total + block - 1) / block);
}

// ---------------- order-preserving float<->uint encoding for atomicMax ----------
__device__ __forceinline__ unsigned enc_f32(float f) {
  unsigned u = __float_as_uint(f);
  return (u & 0x80000000u) ? ~u : (u | 0x80000000u);
}
__device__ __forceinline__ float dec_f32(unsigned e) {
  unsigned u = (e & 0x80000000u) ? (e ^ 0x80000000u) : ~e;
  return __uint_as_float(u);
}

// ---------------- WMMA GEMM: C[N,M] = act( rowscale(A)[N,K] @ W[K,M] + bias + addm )
// f16 inputs staged to LDS pre-swizzled into fragment order, f32 accumulate.
// 256 threads = 8 waves; CT=M/16 col tiles; RT=8/CT row-tile groups; RR row repeats
// per wave (B fragments reused RR times). ROWS = RT*RR*16 rows per block.
// Fragment maps (wave32, v_wmma_f32_16x16x32_f16):
//   A: half hi=lane>=16; a[t<8]=A[row][K0+8*hi+t]; a[t>=8]=A[row][K0+16+8*hi+t-8]
//   B: b[e<8]=W[K0+8*hi+e][col]; b[e>=8]=W[K0+16+8*hi+e-8][col] (same k map as A)
template <int KPAD, int M, int RR, int ACT, bool RS, bool ADD>
__global__ __launch_bounds__(256) void gemm_wmma(
    const float* __restrict__ A, const float* __restrict__ rowscale,
    const float* __restrict__ W, const float* __restrict__ bias,
    const float* __restrict__ addm, float* __restrict__ C, int N, int K)
{
  constexpr int CT    = M / 16;
  constexpr int RT    = 8 / CT;
  constexpr int AROWS = RT * RR * 16;
  constexpr int CH    = KPAD / 32;

  __shared__ __align__(32) _Float16 Bs[KPAD * M];
  __shared__ __align__(32) _Float16 As[AROWS * KPAD];
  __shared__ float bl[M];

  const int tid = threadIdx.x;

  // ---- stage W, swizzled into B-fragment order ----
  for (int i = tid; i < KPAD * M; i += 256) {
    int k = i / M, m = i - k * M;
    float v = (k < K) ? W[(size_t)k * M + m] : 0.0f;
    int chunk = k >> 5, kin = k & 31;
    int hb = (kin >> 3) & 1, e = (kin & 7) + ((kin >> 4) << 3);
    Bs[(((chunk * M) + m) * 2 + hb) * 16 + e] = (_Float16)v;
  }
  for (int m = tid; m < M; m += 256) bl[m] = bias ? bias[m] : 0.0f;

  // ---- stage A row strip, row-scaled, swizzled into A-fragment order ----
  const int r0 = blockIdx.x * AROWS;
  for (int i = tid; i < AROWS * KPAD; i += 256) {
    int r = i / KPAD, k = i - r * KPAD;
    int gr = r0 + r;
    float v = 0.0f;
    if (gr < N && k < K) {
      v = A[(size_t)gr * K + k];
      if (RS) v *= rowscale[gr];
    }
    int chunk = k >> 5, kin = k & 31;
    int ha = (kin >> 3) & 1;
    int t = (kin & 7) + ((kin >> 4) << 3);
    As[((r * CH + chunk) * 2 + ha) * 16 + t] = (_Float16)v;
  }
  __syncthreads();

  const int wave = tid >> 5, lane = tid & 31;
  const int rowTileBase = (wave / CT) * RR;
  const int colTile = wave % CT;
  const int hi  = (lane >= 16);
  const int l16 = lane & 15;
  const int bcol = colTile * 16 + l16;

  v8f acc[RR];
#pragma unroll
  for (int rr = 0; rr < RR; ++rr) acc[rr] = (v8f){};

#pragma unroll
  for (int c = 0; c < CH; ++c) {
    v16h b = *(const v16h*)&Bs[((c * M + bcol) * 2 + hi) * 16];
#pragma unroll
    for (int rr = 0; rr < RR; ++rr) {
      const int arow = (rowTileBase + rr) * 16 + l16;
      v16h a = *(const v16h*)&As[((arow * CH + c) * 2 + hi) * 16];
      acc[rr] = __builtin_amdgcn_wmma_f32_16x16x32_f16(false, a, false, b,
                                                       (short)0, acc[rr], false, false);
    }
  }

  // ---- epilogue: hoisted bias, uniform full/partial split ----
  const float bb = bl[bcol];
  if (r0 + AROWS <= N) {           // full block: straight-line, no per-element guards
#pragma unroll
    for (int rr = 0; rr < RR; ++rr) {
      const int gr0 = r0 + (rowTileBase + rr) * 16 + (hi ? 8 : 0);
#pragma unroll
      for (int v = 0; v < 8; ++v) {
        const int gr = gr0 + v;
        float val = acc[rr][v] + bb;
        if (ADD) val += addm[(size_t)gr * M + bcol];
        if (ACT == 1) val = val > 0.0f ? val : 0.0f;
        C[(size_t)gr * M + bcol] = val;
      }
    }
  } else {                          // tail block
#pragma unroll
    for (int rr = 0; rr < RR; ++rr) {
      const int gr0 = r0 + (rowTileBase + rr) * 16 + (hi ? 8 : 0);
#pragma unroll
      for (int v = 0; v < 8; ++v) {
        const int gr = gr0 + v;
        if (gr < N) {
          float val = acc[rr][v] + bb;
          if (ADD) val += addm[(size_t)gr * M + bcol];
          if (ACT == 1) val = val > 0.0f ? val : 0.0f;
          C[(size_t)gr * M + bcol] = val;
        }
      }
    }
  }
}

// ---------------- edge scatter-add: AGG[dst] += H[src], D a multiple of 4 ------------
__global__ void scatter_add4(const float* __restrict__ H, const int* __restrict__ src,
                             const int* __restrict__ dst, float* __restrict__ AGG,
                             int E, int D4)
{
  int idx = blockIdx.x * blockDim.x + threadIdx.x;
  if (idx >= E * D4) return;
  int e = idx / D4, c = (idx - e * D4) * 4;
  int s = src[e], d = dst[e];
  const float4 v = *(const float4*)(H + (size_t)s * D4 * 4 + c);
  float* p = AGG + (size_t)d * D4 * 4 + c;
  atomicAdd(p + 0, v.x); atomicAdd(p + 1, v.y);
  atomicAdd(p + 2, v.z); atomicAdd(p + 3, v.w);
}

// ---------------- mask = sigmoid([s2 || agg] @ w3 + b3), scalar per node ------------
__global__ void mask_sigmoid(const float* __restrict__ s2, const float* __restrict__ agg,
                             const float* __restrict__ w3, const float* __restrict__ b3,
                             float* __restrict__ mask, int N)
{
  __shared__ float w[256];
  for (int i = threadIdx.x; i < 256; i += blockDim.x) w[i] = w3[i];
  __syncthreads();
  int i = blockIdx.x * blockDim.x + threadIdx.x;
  if (i >= N) return;
  float s = b3[0];
  const float* s2p = s2 + (size_t)i * 128;
  const float* agp = agg + (size_t)i * 128;
#pragma unroll 4
  for (int k = 0; k < 128; ++k) s += s2p[k] * w[k] + agp[k] * w[128 + k];
  mask[i] = 1.0f / (1.0f + __expf(-s));
}

// ---------------- per-(node,head) attention logits ----------------------------------
__global__ void att_score(const float* __restrict__ hmat, const float* __restrict__ att,
                          float* __restrict__ out, int N)
{
  int i = blockIdx.x * blockDim.x + threadIdx.x;
  if (i >= N * 2) return;
  int node = i >> 1, h = i & 1;
  const float* hp = hmat + (size_t)node * 64 + h * 32;
  const float* ap = att + h * 32;
  float s = 0.0f;
#pragma unroll
  for (int d = 0; d < 32; ++d) s += hp[d] * ap[d];
  out[i] = s;
}

// ---------------- GAT edge passes (src==nullptr -> self-loop identity mode) ----------
__global__ void gat_edge_max(const float* __restrict__ as, const float* __restrict__ ad,
                             const int* __restrict__ src, const int* __restrict__ dst,
                             unsigned* __restrict__ mbuf, int E)
{
  int idx = blockIdx.x * blockDim.x + threadIdx.x;
  if (idx >= E * 2) return;
  int e = idx >> 1, h = idx & 1;
  int s = src ? src[e] : e;
  int d = dst ? dst[e] : e;
  float v = as[s * 2 + h] + ad[d * 2 + h];
  v = v > 0.0f ? v : NEG_SLOPE * v;
  atomicMax(&mbuf[d * 2 + h], enc_f32(v));
}

__global__ void gat_edge_exp(const float* __restrict__ as, const float* __restrict__ ad,
                             const int* __restrict__ src, const int* __restrict__ dst,
                             const unsigned* __restrict__ mbuf, float* __restrict__ zbuf,
                             float* __restrict__ alpha, int E)
{
  int idx = blockIdx.x * blockDim.x + threadIdx.x;
  if (idx >= E * 2) return;
  int e = idx >> 1, h = idx & 1;
  int s = src ? src[e] : e;
  int d = dst ? dst[e] : e;
  float v = as[s * 2 + h] + ad[d * 2 + h];
  v = v > 0.0f ? v : NEG_SLOPE * v;
  float w = __expf(v - dec_f32(mbuf[d * 2 + h]));
  alpha[idx] = w;
  atomicAdd(&zbuf[d * 2 + h], w);
}

__global__ void gat_edge_acc(const float* __restrict__ hs, const float* __restrict__ alpha,
                             const int* __restrict__ src, const int* __restrict__ dst,
                             float* __restrict__ outacc, int E)
{
  int idx = blockIdx.x * blockDim.x + threadIdx.x;
  if (idx >= E * 16) return;
  int e = idx >> 4, r = idx & 15;
  int h = r >> 3, c4 = (r & 7) * 4;
  int s = src ? src[e] : e;
  int d = dst ? dst[e] : e;
  float w = alpha[e * 2 + h];
  const float4 v = *(const float4*)(hs + (size_t)s * 64 + h * 32 + c4);
  float* p = outacc + (size_t)d * 64 + h * 32 + c4;
  atomicAdd(p + 0, v.x * w); atomicAdd(p + 1, v.y * w);
  atomicAdd(p + 2, v.z * w); atomicAdd(p + 3, v.w * w);
}

__global__ void gat_final(const float* __restrict__ outacc, const float* __restrict__ zbuf,
                          const float* __restrict__ bias, float* __restrict__ rep,
                          int N, int coff)
{
  int idx = blockIdx.x * blockDim.x + threadIdx.x;
  if (idx >= N * 64) return;
  int i = idx >> 6, c = idx & 63, h = c >> 5;
  float z = zbuf[i * 2 + h];
  float v = (z != 0.0f) ? outacc[idx] / z : 0.0f;
  rep[(size_t)i * 128 + coff + c] = v + bias[c];
}

// ---------------- SAG pooling -------------------------------------------------------
__global__ void sag_score(const float* __restrict__ rep, const float* __restrict__ agg,
                          const float* __restrict__ wroot, const float* __restrict__ wrel,
                          const float* __restrict__ b, float* __restrict__ score, int N)
{
  __shared__ float wr[128], wl[128];
  for (int i = threadIdx.x; i < 128; i += blockDim.x) { wr[i] = wroot[i]; wl[i] = wrel[i]; }
  __syncthreads();
  int i = blockIdx.x * blockDim.x + threadIdx.x;
  if (i >= N) return;
  float s = b[0];
  const float* rp = rep + (size_t)i * 128;
  const float* ap = agg + (size_t)i * 128;
#pragma unroll 4
  for (int k = 0; k < 128; ++k) s += rp[k] * wr[k] + ap[k] * wl[k];
  score[i] = s;
}

__global__ void seg_max_1d(const float* __restrict__ s, const int* __restrict__ seg,
                           unsigned* __restrict__ gm, int N)
{
  int i = blockIdx.x * blockDim.x + threadIdx.x;
  if (i >= N) return;
  atomicMax(&gm[seg[i]], enc_f32(s[i]));
}

__global__ void seg_expsum_1d(float* __restrict__ s, const int* __restrict__ seg,
                              const unsigned* __restrict__ gm, float* __restrict__ gz, int N)
{
  int i = blockIdx.x * blockDim.x + threadIdx.x;
  if (i >= N) return;
  float w = __expf(s[i] - dec_f32(gm[seg[i]]));
  s[i] = w;
  atomicAdd(&gz[seg[i]], w);
}

__global__ void pool_acc4(const float* __restrict__ rep, const float* __restrict__ w,
                          const int* __restrict__ seg, float* __restrict__ emb, int N)
{
  int idx = blockIdx.x * blockDim.x + threadIdx.x;
  if (idx >= N * 32) return;
  int i = idx >> 5, c4 = (idx & 31) * 4;
  int g = seg[i];
  float ww = w[i];
  const float4 v = *(const float4*)(rep + (size_t)i * 128 + c4);
  float* p = emb + (size_t)g * 128 + c4;
  atomicAdd(p + 0, v.x * ww); atomicAdd(p + 1, v.y * ww);
  atomicAdd(p + 2, v.z * ww); atomicAdd(p + 3, v.w * ww);
}

__global__ void pool_norm(float* __restrict__ emb, const float* __restrict__ gz)
{
  int idx = blockIdx.x * blockDim.x + threadIdx.x;
  if (idx >= NGRAPHS * 128) return;
  float z = gz[idx >> 7];
  emb[idx] = (z != 0.0f) ? emb[idx] / z : 0.0f;
}

// ====================================================================================
extern "C" void kernel_launch(void* const* d_in, const int* in_sizes, int n_in,
                              void* d_out, int out_size, void* d_ws, size_t ws_size,
                              hipStream_t stream)
{
  const float* h_x   = (const float*)d_in[0];
  const float* t_x   = (const float*)d_in[1];
  const int*   he    = (const int*)d_in[2];
  const int*   te    = (const int*)d_in[3];
  const int*   be    = (const int*)d_in[4];
  const int*   h_bat = (const int*)d_in[5];
  const int*   t_bat = (const int*)d_in[6];
  const float* lin0_w = (const float*)d_in[7];
  const float* lin0_b = (const float*)d_in[8];
  const float* mw1 = (const float*)d_in[9];   const float* mb1 = (const float*)d_in[10];
  const float* mw2 = (const float*)d_in[11];  const float* mb2 = (const float*)d_in[12];
  const float* mw3 = (const float*)d_in[13];  const float* mb3 = (const float*)d_in[14];
  const float* cw1 = (const float*)d_in[15];  const float* cb1 = (const float*)d_in[16];
  const float* cw2 = (const float*)d_in[17];  const float* cb2 = (const float*)d_in[18];
  const float* intra_w  = (const float*)d_in[19];
  const float* intra_as = (const float*)d_in[20];
  const float* intra_ad = (const float*)d_in[21];
  const float* intra_b  = (const float*)d_in[22];
  const float* inter_ws = (const float*)d_in[23];
  const float* inter_wd = (const float*)d_in[24];
  const float* inter_as = (const float*)d_in[25];
  const float* inter_ad = (const float*)d_in[26];
  const float* inter_b  = (const float*)d_in[27];
  const float* sag_wroot = (const float*)d_in[28];
  const float* sag_wrel  = (const float*)d_in[29];
  const float* sag_bb    = (const float*)d_in[30];

  const int Nh = in_sizes[0] / INFEAT, Nt = in_sizes[1] / INFEAT;
  const int Eh = in_sizes[2] / 2, Et = in_sizes[3] / 2, Eb = in_sizes[4] / 2;
  const int Nmax = Nh > Nt ? Nh : Nt;
  int Emax = Eh; if (Et > Emax) Emax = Et; if (Eb > Emax) Emax = Eb;

  const int *hsrc = he, *hdst = he + Eh;
  const int *tsrc = te, *tdst = te + Et;
  const int *bsrc = be, *bdst = be + Eb;

  // ---- workspace carving (256B aligned) ----
  size_t off = 0;
  auto alloc = [&](size_t elems) -> float* {
    float* p = (float*)((char*)d_ws + off);
    off += (elems * sizeof(float) + 255) & ~(size_t)255;
    return p;
  };
  float* xh    = alloc((size_t)Nh * 128);
  float* xt    = alloc((size_t)Nt * 128);
  float* rep_h = alloc((size_t)Nh * 128);
  float* rep_t = alloc((size_t)Nt * 128);
  float* P0 = alloc((size_t)Nmax * 128);
  float* P1 = alloc((size_t)Nmax * 128);
  float* P2 = alloc((size_t)Nmax * 128);
  float* P3 = alloc((size_t)Nmax * 128);
  float* maskA = alloc(Nmax);
  float* maskB = alloc(Nmax);
  float* S1 = alloc((size_t)Nmax * 64);
  float* S2 = alloc((size_t)Nmax * 64);
  float* a_src = alloc((size_t)Nmax * 2);
  float* a_dst = alloc((size_t)Nmax * 2);
  unsigned* mbuf = (unsigned*)alloc((size_t)Nmax * 2);
  float* zbuf   = alloc((size_t)Nmax * 2);
  float* alphaE = alloc((size_t)Emax * 2);
  float* alphaS = alloc((size_t)Nmax * 2);
  float* outacc = alloc((size_t)Nmax * 64);
  unsigned* gm = (unsigned*)alloc(NGRAPHS);
  float* gz = alloc(NGRAPHS);
  (void)ws_size; (void)n_in; (void)out_size;

  // GEMM dispatch over the 5 used configurations (ROWS per block = 64 for all)
  // cfg: 0=lin0(K70,M128), 1=plain(K128,M128), 2=rowscaled(K128,M128),
  //      3=rowscale+add+relu(K128,M128), 4=gat(K128,M64, no bias)
  auto gemm = [&](int cfg, const float* A, const float* rs, const float* W,
                  const float* bias, const float* addm, float* C, int N, int K) {
    if (cfg == 4) {
      gemm_wmma<128, 64, 2, 0, false, false>
          <<<gdim(N, 64), 256, 0, stream>>>(A, rs, W, bias, addm, C, N, K);
    } else if (cfg == 0) {
      gemm_wmma<96, 128, 4, 0, false, false>
          <<<gdim(N, 64), 256, 0, stream>>>(A, rs, W, bias, addm, C, N, K);
    } else if (cfg == 1) {
      gemm_wmma<128, 128, 4, 0, false, false>
          <<<gdim(N, 64), 256, 0, stream>>>(A, rs, W, bias, addm, C, N, K);
    } else if (cfg == 2) {
      gemm_wmma<128, 128, 4, 0, true, false>
          <<<gdim(N, 64), 256, 0, stream>>>(A, rs, W, bias, addm, C, N, K);
    } else {
      gemm_wmma<128, 128, 4, 1, true, true>
          <<<gdim(N, 64), 256, 0, stream>>>(A, rs, W, bias, addm, C, N, K);
    }
  };
  auto scatter = [&](const float* H, const int* s, const int* d, float* AGG,
                     int Nn, int E, int D) {
    hipMemsetAsync(AGG, 0, (size_t)Nn * D * sizeof(float), stream);
    scatter_add4<<<gdim((long long)E * (D / 4), 256), 256, 0, stream>>>(H, s, d, AGG, E, D / 4);
  };

  // ---- branch: lin0 + 3x (WeightConv mask; SparseConv); elu(relu(x)) == relu(x) ----
  auto branch = [&](const float* x0, int N, int E, const int* s, const int* d, float* xout) {
    float *cur = P0, *b1 = P1, *b2 = P2, *b3 = P3;
    float *mprev = maskA, *mcur = maskB;
    gemm(0, x0, nullptr, lin0_w, lin0_b, nullptr, cur, N, INFEAT);
    for (int l = 0; l < 3; ++l) {
      const float* rs = (l == 0) ? nullptr : mprev;
      int cfgw = (l == 0) ? 1 : 2;
      gemm(cfgw, cur, rs, mw1 + (size_t)l * 128 * 128, mb1 + l * 128, nullptr, b1, N, 128);
      scatter(b1, s, d, b2, N, E, 128);                                  // aggm
      gemm(cfgw, cur, rs, mw2 + (size_t)l * 128 * 128, mb2 + l * 128, nullptr, b3, N, 128);
      mask_sigmoid<<<gdim(N, 256), 256, 0, stream>>>(b3, b2, mw3 + l * 256, mb3 + l, mcur, N);
      gemm(2, cur, mcur, cw1 + (size_t)l * 128 * 128, cb1 + l * 128, nullptr, b1, N, 128);
      scatter(b1, s, d, b2, N, E, 128);                                  // aggc
      float* nxt = (l == 2) ? xout : b3;
      gemm(3, cur, mcur, cw2 + (size_t)l * 128 * 128, cb2 + l * 128, b2, nxt, N, 128);
      if (l < 2) { float* t = cur; cur = b3; b3 = t; }
      { float* t = mprev; mprev = mcur; mcur = t; }
    }
  };

  // ---- GAT (selfloop==true adds identity-edge passes; requires Ns==Nd) -------------
  auto gat = [&](const float* xs, const float* xd, int Ns, int Nd,
                 const int* s, const int* d, int E, bool selfloop,
                 const float* Ws, const float* Wd, const float* attS, const float* attD,
                 const float* bias, float* rep, int coff) {
    gemm(4, xs, nullptr, Ws, nullptr, nullptr, S1, Ns, 128);
    const float* hd = S1;
    if (!(xs == xd && Ws == Wd)) {
      gemm(4, xd, nullptr, Wd, nullptr, nullptr, S2, Nd, 128);
      hd = S2;
    }
    att_score<<<gdim((long long)Ns * 2, 256), 256, 0, stream>>>(S1, attS, a_src, Ns);
    att_score<<<gdim((long long)Nd * 2, 256), 256, 0, stream>>>(hd, attD, a_dst, Nd);
    hipMemsetAsync(mbuf, 0, (size_t)Nd * 2 * sizeof(unsigned), stream);
    hipMemsetAsync(zbuf, 0, (size_t)Nd * 2 * sizeof(float), stream);
    hipMemsetAsync(outacc, 0, (size_t)Nd * 64 * sizeof(float), stream);
    gat_edge_max<<<gdim((long long)E * 2, 256), 256, 0, stream>>>(a_src, a_dst, s, d, mbuf, E);
    if (selfloop)
      gat_edge_max<<<gdim((long long)Nd * 2, 256), 256, 0, stream>>>(a_src, a_dst, nullptr, nullptr, mbuf, Nd);
    gat_edge_exp<<<gdim((long long)E * 2, 256), 256, 0, stream>>>(a_src, a_dst, s, d, mbuf, zbuf, alphaE, E);
    if (selfloop)
      gat_edge_exp<<<gdim((long long)Nd * 2, 256), 256, 0, stream>>>(a_src, a_dst, nullptr, nullptr, mbuf, zbuf, alphaS, Nd);
    gat_edge_acc<<<gdim((long long)E * 16, 256), 256, 0, stream>>>(S1, alphaE, s, d, outacc, E);
    if (selfloop)
      gat_edge_acc<<<gdim((long long)Nd * 16, 256), 256, 0, stream>>>(S1, alphaS, nullptr, nullptr, outacc, Nd);
    gat_final<<<gdim((long long)Nd * 64, 256), 256, 0, stream>>>(outacc, zbuf, bias, rep, Nd, coff);
  };

  // ---- SAG pool + global add ------------------------------------------------------
  auto pool = [&](const float* rep, int N, const int* s, const int* d, int E,
                  const int* batch, float* emb) {
    scatter(rep, s, d, P0, N, E, 128);
    sag_score<<<gdim(N, 256), 256, 0, stream>>>(rep, P0, sag_wroot, sag_wrel, sag_bb, maskA, N);
    hipMemsetAsync(gm, 0, NGRAPHS * sizeof(unsigned), stream);
    hipMemsetAsync(gz, 0, NGRAPHS * sizeof(float), stream);
    seg_max_1d<<<gdim(N, 256), 256, 0, stream>>>(maskA, batch, gm, N);
    seg_expsum_1d<<<gdim(N, 256), 256, 0, stream>>>(maskA, batch, gm, gz, N);
    hipMemsetAsync(emb, 0, (size_t)NGRAPHS * 128 * sizeof(float), stream);
    pool_acc4<<<gdim((long long)N * 32, 256), 256, 0, stream>>>(rep, maskA, batch, emb, N);
    pool_norm<<<gdim(NGRAPHS * 128, 256), 256, 0, stream>>>(emb, gz);
  };

  // ================= full forward =================
  branch(h_x, Nh, Eh, hsrc, hdst, xh);
  branch(t_x, Nt, Et, tsrc, tdst, xt);
  // elu(relu(x)) == relu(x): xh/xt are already the ELU'd representations.

  gat(xh, xh, Nh, Nh, hsrc, hdst, Eh, true,  intra_w,  intra_w,  intra_as, intra_ad, intra_b, rep_h, 0);
  gat(xt, xt, Nt, Nt, tsrc, tdst, Et, true,  intra_w,  intra_w,  intra_as, intra_ad, intra_b, rep_t, 0);
  gat(xh, xt, Nh, Nt, bsrc, bdst, Eb, false, inter_ws, inter_wd, inter_as, inter_ad, inter_b, rep_t, 64);
  gat(xt, xh, Nt, Nh, bdst, bsrc, Eb, false, inter_ws, inter_wd, inter_as, inter_ad, inter_b, rep_h, 64);

  float* out = (float*)d_out;
  pool(rep_h, Nh, hsrc, hdst, Eh, h_bat, out);
  pool(rep_t, Nt, tsrc, tdst, Et, t_bat, out + NGRAPHS * 128);
}
